// Simulator_42743514530226
// MI455X (gfx1250) — matrix operand order repaired and checked by reading, not compiled, not run on hardware
//
#include <hip/hip_runtime.h>
#include <hip/hip_bf16.h>

// Problem dimensions (match reference).
#define N_X   1024
#define N_U   512
#define N_Y   128
#define HID   4096
#define T_STEPS 8192
#define RECUR_BLOCKS 128

typedef __attribute__((ext_vector_type(16))) __bf16        v16bf;
typedef __attribute__((ext_vector_type(8)))  float         v8f;
typedef __attribute__((ext_vector_type(8)))  unsigned int  v8u;

union FragU { v8u u; v16bf b; };
union FragQ { uint4 q[2]; v16bf b; };

// ---------- small helpers ----------
__device__ __forceinline__ unsigned short f2bf(float f) {
  unsigned int u = __float_as_uint(f);
  u += 0x7FFFu + ((u >> 16) & 1u);          // round-to-nearest-even
  return (unsigned short)(u >> 16);
}
__device__ __forceinline__ float bf2f(unsigned short v) {
  return __uint_as_float(((unsigned int)v) << 16);
}
__device__ __forceinline__ float bflo(unsigned int u) { return __uint_as_float(u << 16); }
__device__ __forceinline__ float bfhi(unsigned int u) { return __uint_as_float(u & 0xFFFF0000u); }

__device__ __forceinline__ v8f wmma_bf16(v16bf a, v16bf b, v8f c) {
  // D = A(16x32 bf16) * B(32x16 bf16) + C(16x16 f32)
  return __builtin_amdgcn_wmma_f32_16x16x32_bf16(
      /*neg_a=*/false, a, /*neg_b=*/false, b,
      /*c_mod=*/(short)0, c, /*reuse_a=*/false, /*reuse_b=*/false);
}

// CDNA5 async copy: 16B global -> LDS, tracked by ASYNCcnt (per ISA 08_async_tensor).
__device__ __forceinline__ void async_ld_b128(unsigned lds_off, const void* gaddr) {
  asm volatile("global_load_async_to_lds_b128 %0, %1, off"
               :: "v"(lds_off), "v"(gaddr) : "memory");
}
__device__ __forceinline__ void wait_async0() {
  asm volatile("s_wait_asynccnt 0x0" ::: "memory");
}
// Low 32 bits of a generic pointer to a __shared__ object are the raw LDS offset
// (ISA 10.2: LDS aperture address = {SHARED_BASE, offset[31:0]}).
__device__ __forceinline__ unsigned lds_off_of(const void* p) {
  return (unsigned)(unsigned long long)p;
}

// ---------- fragment loaders ----------
// A fragment from global row-major bf16 [M, ld], tile at (m0, k0).
// ISA layout (16-bit A 16x32): lanes 0-15 -> M rows, half=lane>>4;
// VGPR g<4: K = half*8+2g ; g>=4: K = 16+half*8+2(g-4).
__device__ __forceinline__ v16bf load_frag_A(const unsigned short* __restrict__ A,
                                             int ld, int m0, int k0, int lane) {
  const int m    = m0 + (lane & 15);
  const int half = lane >> 4;
  const unsigned short* row = A + (long)m * ld + k0;
  FragU f;
#pragma unroll
  for (int g = 0; g < 8; ++g) {
    const int koff = (g < 4) ? (half * 8 + 2 * g) : (16 + half * 8 + 2 * (g - 4));
    f.u[g] = *(const unsigned int*)(row + koff);
  }
  return f.b;
}

// B fragment from pre-transposed storage BT[N, ld] (K contiguous per n row).
// Same K layout as A: VGPR g<4: K = half*8+2g ; g>=4: K = 16+half*8+2(g-4).
__device__ __forceinline__ v16bf load_frag_BT(const unsigned short* __restrict__ BT,
                                              int ld, int n0, int k0, int lane) {
  const int n    = n0 + (lane & 15);
  const int half = lane >> 4;
  const unsigned short* row = BT + (long)n * ld + k0;
  FragU f;
#pragma unroll
  for (int g = 0; g < 8; ++g) f.u[g] = *(const unsigned int*)(row + ((g < 4) ? (half * 8 + 2 * g) : (16 + half * 8 + 2 * (g - 4))));
  return f.b;
}

// Same fragments out of an LDS-staged K-slab (row-major [rows][32] bf16):
// each half-lane's K-span is contiguous -> two 16B ds reads per fragment.
__device__ __forceinline__ v16bf lds_frag_A(const unsigned short* At, int row, int half) {
  const unsigned short* rp = At + row * 32;
  FragQ f;
  f.q[0] = *(const uint4*)(rp + half * 8);        // K = half*8   .. half*8+7
  f.q[1] = *(const uint4*)(rp + 16 + half * 8);   // K = 16+half*8 ..
  return f.b;
}
__device__ __forceinline__ v16bf lds_frag_B(const unsigned short* Bt, int n, int half) {
  const unsigned short* rp = Bt + n * 32;   // K = half*8.., 16+half*8..
  FragQ f;
  f.q[0] = *(const uint4*)(rp + half * 8);
  f.q[1] = *(const uint4*)(rp + 16 + half * 8);
  return f.b;
}

// ---------- conversion kernels ----------
__global__ void cvt_kernel(const float* __restrict__ src, unsigned short* __restrict__ dst, long n) {
  long i = (long)blockIdx.x * blockDim.x + threadIdx.x;
  if (i < n) dst[i] = f2bf(src[i]);
}

// dst[c*R + r] = src[(row0+r)*srcLd + c]   (convert + transpose; dst writes coalesced)
__global__ void cvtT_kernel(const float* __restrict__ src, unsigned short* __restrict__ dst,
                            int R, int C, int row0, int srcLd) {
  long i = (long)blockIdx.x * blockDim.x + threadIdx.x;
  long total = (long)R * C;
  if (i >= total) return;
  int r = (int)(i % R);
  int c = (int)(i / R);
  dst[i] = f2bf(src[(long)(row0 + r) * srcLd + c]);
}

// X[0]=x0 ; X[t]=f_b2 broadcast for t>=1 (atomic layer-2 accumulates on top of bias).
__global__ void initX_kernel(const float* __restrict__ x0, const float* __restrict__ f_b2,
                             float* __restrict__ X, unsigned int* __restrict__ bar) {
  long i = (long)blockIdx.x * blockDim.x + threadIdx.x;
  if (i == 0) { bar[0] = 0u; bar[1] = 0u; }
  if (i >= (long)T_STEPS * N_X) return;
  int t = (int)(i / N_X);
  int j = (int)(i % N_X);
  X[i] = (t == 0) ? x0[j] : f_b2[j];
}

// ---------- tiled WMMA GEMM: 256M x 128N macro-tile, async double-buffered LDS ----------
// out = EPI==0 ?  A@B^T + bias  (f32)   :   bf16(tanh(A@B^T + bias))
// Waves arranged 4(y) x 2(x); each wave: 4x4 = 16 accumulator fragments, 16 WMMA / K-step.
template <int EPI>
__global__ void __launch_bounds__(256)
wmma_tile_kernel(const unsigned short* __restrict__ A, int ldA,
                 const unsigned short* __restrict__ BT, int ldB, int kdim,
                 const float* __restrict__ bias, void* __restrict__ outv, int ldOut) {
  __shared__ unsigned short Atile[2][256 * 32];   // 2 x 16KB
  __shared__ unsigned short Btile[2][128 * 32];   // 2 x  8KB
  const int tid  = threadIdx.x;
  const int lane = tid & 31, wave = tid >> 5;
  const int wy = wave >> 1, wx = wave & 1;
  const int half = lane >> 4, nc = lane & 15;
  const int m0 = blockIdx.y * 256;
  const int n0 = blockIdx.x * 128;

  v8f acc[4][4];
#pragma unroll
  for (int ty = 0; ty < 4; ++ty)
#pragma unroll
    for (int nb = 0; nb < 4; ++nb) acc[ty][nb] = (v8f)0.0f;

  // Stage one 32-wide K-slab of A (256 rows) and B^T (128 rows) into LDS.
  auto stageA = [&](int buf, int k0) {
    const unsigned short* g = A + ((long)(m0 + tid) * ldA + k0);   // row = tid
    unsigned l = lds_off_of(&Atile[buf][tid * 32]);
#pragma unroll
    for (int seg = 0; seg < 4; ++seg)
      async_ld_b128(l + seg * 16, (const char*)g + seg * 16);
  };
  auto stageB = [&](int buf, int k0) {
#pragma unroll
    for (int j = 0; j < 2; ++j) {
      const int idx = tid + 256 * j;          // 0..511
      const int n = idx >> 2, seg = idx & 3;
      const unsigned short* g = BT + ((long)(n0 + n) * ldB + k0);
      unsigned l = lds_off_of(&Btile[buf][n * 32]) + seg * 16;
      async_ld_b128(l, (const char*)g + seg * 16);
    }
  };

  const int nK = kdim / 32;
  stageA(0, 0);
  stageB(0, 0);
  for (int kk = 0; kk < nK; ++kk) {
    wait_async0();        // my async writes for buf[kk&1] have landed
    __syncthreads();      // everyone's have
    if (kk + 1 < nK) {    // prefetch next slab into the other buffer
      stageA((kk + 1) & 1, (kk + 1) * 32);
      stageB((kk + 1) & 1, (kk + 1) * 32);
    }
    const unsigned short* At = &Atile[kk & 1][0];
    const unsigned short* Bt = &Btile[kk & 1][0];
    v16bf af[4];
#pragma unroll
    for (int ty = 0; ty < 4; ++ty)
      af[ty] = lds_frag_A(At, wy * 64 + ty * 16 + nc, half);
#pragma unroll
    for (int nb = 0; nb < 4; ++nb) {
      v16bf bfv = lds_frag_B(Bt, wx * 64 + nb * 16 + nc, half);
#pragma unroll
      for (int ty = 0; ty < 4; ++ty)
        acc[ty][nb] = wmma_bf16(af[ty], bfv, acc[ty][nb]);
    }
    // no trailing barrier needed: next iteration's wait+barrier protects reuse
  }

#pragma unroll
  for (int ty = 0; ty < 4; ++ty)
#pragma unroll
    for (int nb = 0; nb < 4; ++nb)
#pragma unroll
      for (int r = 0; r < 8; ++r) {
        const int m = m0 + wy * 64 + ty * 16 + r + 8 * half;
        const int n = n0 + wx * 64 + nb * 16 + nc;
        const float v = acc[ty][nb][r] + bias[n];
        if (EPI == 0) ((float*)outv)[(long)m * ldOut + n] = v;
        else          ((unsigned short*)outv)[(long)m * ldOut + n] = f2bf(tanhf(v));
      }
}

// Y[T,N_Y] = H[T,HID] @ g_w2 + g_b2   (B stored transposed [N_Y,HID]); wave: 16 rows x 128 cols
__global__ void __launch_bounds__(256)
wmma_y_kernel(const unsigned short* __restrict__ H,
              const unsigned short* __restrict__ W,         // [N_Y, HID]
              const float* __restrict__ bias,               // [N_Y]
              float* __restrict__ out) {                    // [T, N_Y]
  const int lane = threadIdx.x & 31, wave = threadIdx.x >> 5;
  const int m0 = blockIdx.x * 128 + wave * 16;
  v8f acc[8];
#pragma unroll
  for (int nb = 0; nb < 8; ++nb) acc[nb] = (v8f)0.0f;
  for (int k0 = 0; k0 < HID; k0 += 32) {
    v16bf a = load_frag_A(H, HID, m0, k0, lane);
#pragma unroll
    for (int nb = 0; nb < 8; ++nb) {
      v16bf b = load_frag_BT(W, HID, nb * 16, k0, lane);
      acc[nb] = wmma_bf16(a, b, acc[nb]);
    }
  }
  const int half = lane >> 4, nc = lane & 15;
#pragma unroll
  for (int nb = 0; nb < 8; ++nb)
#pragma unroll
    for (int r = 0; r < 8; ++r) {
      int m = m0 + r + 8 * half;
      int n = nb * 16 + nc;
      out[(long)m * N_Y + n] = acc[nb][r] + bias[n];
    }
}

// ---------- persistent sequential recurrence ----------
__device__ __forceinline__ void grid_barrier(unsigned int* bar, unsigned int nblk) {
  __threadfence();
  __syncthreads();
  if (threadIdx.x == 0) {
    unsigned int g = __hip_atomic_load(&bar[1], __ATOMIC_ACQUIRE, __HIP_MEMORY_SCOPE_AGENT);
    unsigned int a = __hip_atomic_fetch_add(&bar[0], 1u, __ATOMIC_ACQ_REL, __HIP_MEMORY_SCOPE_AGENT);
    if (a == nblk - 1u) {
      __hip_atomic_store(&bar[0], 0u, __ATOMIC_RELAXED, __HIP_MEMORY_SCOPE_AGENT);
      __hip_atomic_fetch_add(&bar[1], 1u, __ATOMIC_RELEASE, __HIP_MEMORY_SCOPE_AGENT);
    } else {
      while (__hip_atomic_load(&bar[1], __ATOMIC_ACQUIRE, __HIP_MEMORY_SCOPE_AGENT) == g) {
        __builtin_amdgcn_s_sleep(2);
      }
    }
  }
  __syncthreads();
}

// x_{t+1} = tanh(x_t @ f_w1_x + Ubias[t]) @ f_w2  (+ f_b2 pre-seeded in X[t+1]).
// Block b owns hidden neurons [b*32, b*32+32); one grid barrier per step.
__global__ void __launch_bounds__(256)
recur_kernel(float* __restrict__ X,                       // [T, N_X] fp32 states
             const float* __restrict__ Ubias,             // [T, HID] (includes f_b1)
             const unsigned int* __restrict__ fw1xT,      // [HID, N_X] bf16, viewed as uint pairs
             const unsigned short* __restrict__ fw2,      // [HID, N_X] bf16 (original orientation)
             unsigned int* __restrict__ bar) {
  __shared__ float hsh[32];
  const int tid = threadIdx.x;
  const int wave = tid >> 5, lane = tid & 31;
  const int b = blockIdx.x;

  for (int t = 0; t < T_STEPS - 1; ++t) {
    const float* xt = X + (long)t * N_X;

    // ---- layer 1: 4 hidden neurons per wave, lanes split K ----
#pragma unroll
    for (int q = 0; q < 4; ++q) {
      const int n = b * 32 + wave * 4 + q;
      const unsigned int* wrow = fw1xT + (long)n * (N_X / 2);
      float s = 0.0f;
#pragma unroll
      for (int i = 0; i < N_X / 64; ++i) {          // 16 iterations, coalesced
        const int p = lane + 32 * i;                // bf16-pair index
        const unsigned int wp = wrow[p];
        const float2 xp = *(const float2*)(xt + 2 * p);
        s += bflo(wp) * xp.x + bfhi(wp) * xp.y;
      }
#pragma unroll
      for (int off = 16; off > 0; off >>= 1) s += __shfl_down(s, off, 32);
      if (lane == 0) hsh[wave * 4 + q] = tanhf(s + Ubias[(long)t * HID + n]);
    }
    __syncthreads();

    // ---- layer 2: rank-32 update scattered into X[t+1] via fp32 atomics ----
    float acc[4] = {0.f, 0.f, 0.f, 0.f};
    const int base = b * 32;
    for (int nn = 0; nn < 32; ++nn) {
      const float h = hsh[nn];
      const unsigned short* wr = fw2 + (long)(base + nn) * N_X;
#pragma unroll
      for (int jj = 0; jj < 4; ++jj) acc[jj] += h * bf2f(wr[jj * 256 + tid]);
    }
    float* xnext = X + (long)(t + 1) * N_X;
#pragma unroll
    for (int jj = 0; jj < 4; ++jj) atomicAdd(xnext + jj * 256 + tid, acc[jj]);

    grid_barrier(bar, RECUR_BLOCKS);
  }
}

// ---------- host launcher ----------
extern "C" void kernel_launch(void* const* d_in, const int* in_sizes, int n_in,
                              void* d_out, int out_size, void* d_ws, size_t ws_size,
                              hipStream_t stream) {
  (void)in_sizes; (void)n_in; (void)out_size; (void)ws_size;

  const float* x0   = (const float*)d_in[0];
  const float* u    = (const float*)d_in[1];
  const float* f_w1 = (const float*)d_in[2];
  const float* f_b1 = (const float*)d_in[3];
  const float* f_w2 = (const float*)d_in[4];
  const float* f_b2 = (const float*)d_in[5];
  const float* g_w1 = (const float*)d_in[6];
  const float* g_b1 = (const float*)d_in[7];
  const float* g_w2 = (const float*)d_in[8];
  const float* g_b2 = (const float*)d_in[9];
  float* y_out = (float*)d_out;

  // ws layout (bytes, 256-aligned)
  char* base = (char*)d_ws;
  unsigned short* Ubf   = (unsigned short*)(base + 0);             //  8,388,608  [T,N_U]
  unsigned short* fw1uT = (unsigned short*)(base + 8388608);       //  4,194,304  [HID,N_U]
  unsigned short* fw1xT = (unsigned short*)(base + 12582912);      //  8,388,608  [HID,N_X]
  unsigned short* fw2bf = (unsigned short*)(base + 20971520);      //  8,388,608  [HID,N_X]
  unsigned short* gw1T  = (unsigned short*)(base + 29360128);      //  8,388,608  [HID,N_X]
  unsigned short* gw2T  = (unsigned short*)(base + 37748736);      //  1,048,576  [N_Y,HID]
  unsigned int*   bar   = (unsigned int*)(base + 38797312);        //        256
  float*          X     = (float*)(base + 38797568);               // 33,554,432  [T,N_X]
  unsigned short* Xbf   = (unsigned short*)(base + 72352000);      // 16,777,216  [T,N_X]
  unsigned short* Hbuf  = (unsigned short*)(base + 89129216);      // 67,108,864  [T,HID]
  float*          Ubias = (float*)(base + 156238080);              //134,217,728  [T,HID]

  auto g1 = [](long n) { return dim3((unsigned)((n + 255) / 256)); };

  // 1) bf16 conversions (+transposes so WMMA B fragments are K-contiguous)
  cvt_kernel <<<g1((long)T_STEPS * N_U), 256, 0, stream>>>(u, Ubf, (long)T_STEPS * N_U);
  cvtT_kernel<<<g1((long)N_U * HID), 256, 0, stream>>>(f_w1, fw1uT, N_U, HID, N_X, HID);
  cvtT_kernel<<<g1((long)N_X * HID), 256, 0, stream>>>(f_w1, fw1xT, N_X, HID, 0, HID);
  cvt_kernel <<<g1((long)HID * N_X), 256, 0, stream>>>(f_w2, fw2bf, (long)HID * N_X);
  cvtT_kernel<<<g1((long)N_X * HID), 256, 0, stream>>>(g_w1, gw1T, N_X, HID, 0, HID);
  cvtT_kernel<<<g1((long)HID * N_Y), 256, 0, stream>>>(g_w2, gw2T, HID, N_Y, 0, N_Y);

  // 2) state buffer init (X[0]=x0, X[t>=1]=f_b2) + barrier reset
  initX_kernel<<<g1((long)T_STEPS * N_X), 256, 0, stream>>>(x0, f_b2, X, bar);

  // 3) Ubias = U @ f_w1_u + f_b1 (dense WMMA GEMM, hoisted out of the recurrence)
  wmma_tile_kernel<0><<<dim3(HID / 128, T_STEPS / 256), 256, 0, stream>>>(
      Ubf, N_U, fw1uT, N_U, N_U, f_b1, (void*)Ubias, HID);

  // 4) sequential recurrence (persistent kernel, 1 grid barrier per step)
  recur_kernel<<<RECUR_BLOCKS, 256, 0, stream>>>(X, Ubias, (const unsigned int*)fw1xT, fw2bf, bar);

  // 5) output path as dense WMMA GEMMs over all recorded states
  cvt_kernel<<<g1((long)T_STEPS * N_X), 256, 0, stream>>>(X, Xbf, (long)T_STEPS * N_X);
  wmma_tile_kernel<1><<<dim3(HID / 128, T_STEPS / 256), 256, 0, stream>>>(
      Xbf, N_X, gw1T, N_X, N_X, g_b1, (void*)Hbuf, HID);
  wmma_y_kernel<<<dim3(T_STEPS / 128), 256, 0, stream>>>(Hbuf, gw2T, g_b2, y_out);
}
